// DeltaEdgeModel_75617194213673
// MI455X (gfx1250) — compile-verified, matrix operand order checked by reading, not run on hardware
//
#include <hip/hip_runtime.h>
#include <hip/hip_bf16.h>

#define E_EDGES 4096
#define DEDGE   256
#define NHEADS  4
#define HDIM    64
#define DNODE   128
#define NCLS    16

typedef __attribute__((ext_vector_type(16))) _Float16     v16h;
typedef __attribute__((ext_vector_type(8)))  float        v8f;
typedef __attribute__((ext_vector_type(8)))  unsigned int v8u;

// ---------------------------------------------------------------------------
// Load a 16x32 f16 fragment (A-layout; also used for B via transposed [N][K]
// storage) from row-major memory with row stride `stride` (elements).
// ISA layout: lanes 0-15 hold rows 0-15, VGPR r<4 -> K=2r,2r+1 ; r>=4 ->
// K=16+2(r-4)... ; lanes 16-31 same rows with K offset +8.
// Consecutive K pairs are contiguous -> b32/b128 loads.
// ---------------------------------------------------------------------------
static __device__ __forceinline__ v16h load_frag16x32(const _Float16* base, int stride) {
  const int lane = threadIdx.x & 31;
  const int row  = lane & 15;
  const int half = lane >> 4;
  const _Float16* p = base + row * stride + half * 8;
  v8u u;
#pragma unroll
  for (int r = 0; r < 8; ++r) {
    const int k = (r < 4) ? (2 * r) : (8 + 2 * r);
    u[r] = *(const unsigned int*)(p + k);
  }
  return __builtin_bit_cast(v16h, u);
}

static __device__ __forceinline__ v8f wmma_f16(v16h a, v16h b, v8f c) {
  // D = A(16x32 f16) * B(32x16 f16) + C(16x16 f32)
  return __builtin_amdgcn_wmma_f32_16x16x32_f16(false, a, false, b, (short)0, c,
                                                false, false);
}

// ---------------------------------------------------------------------------
// Weight transpose + f32->f16:  W[K][N] (f32) -> Bt[N][K] (f16)
// ---------------------------------------------------------------------------
__global__ void transpose_w_kernel(const float* __restrict__ W,
                                   _Float16* __restrict__ Bt, int K, int N) {
  int i = blockIdx.x * blockDim.x + threadIdx.x;
  if (i >= K * N) return;
  int n = i / K, k = i - n * K;
  Bt[i] = (_Float16)W[k * N + n];
}

// ---------------------------------------------------------------------------
// xcat[e][0:128] = nf[src[e]], xcat[e][128:256] = nf[dst[e]]  (f16)
// ---------------------------------------------------------------------------
__global__ void gather_concat_kernel(const float* __restrict__ nf,
                                     const int* __restrict__ srcIdx,
                                     const int* __restrict__ dstIdx,
                                     _Float16* __restrict__ xcat) {
  int i = blockIdx.x * blockDim.x + threadIdx.x;     // E * 256 threads
  int e = i >> 8;
  int j = i & 255;
  int node = (j < DNODE) ? srcIdx[e] : dstIdx[e];
  xcat[i] = (_Float16)nf[node * DNODE + (j & (DNODE - 1))];
}

// ---------------------------------------------------------------------------
// WMMA GEMM:  C[M,N] = A[M,K](f16) * Bt[N,K](f16)^T + bias (+addsrc)
// One wave computes a 16 x (NT*16) tile (A fragment reused across NT WMMAs).
// N, K, NT are compile-time: the WMMA loop has no branches at all.
// Optional fused GELU; outputs optionally f32, f16 row-major, f16 transposed.
// ---------------------------------------------------------------------------
template <int N, int K, int NT>
__global__ __launch_bounds__(128) void gemm_wmma_kernel(
    const _Float16* __restrict__ A, const _Float16* __restrict__ Bt,
    const float* __restrict__ bias, const float* __restrict__ addsrc,
    float* __restrict__ outF32, _Float16* __restrict__ outF16,
    _Float16* __restrict__ outF16T, int M, int do_gelu) {
  constexpr int tilesN = (N + 63) / 64;
  const int wave   = (blockIdx.x * blockDim.x + threadIdx.x) >> 5;
  const int nWaves = (M >> 4) * tilesN;
  if (wave >= nWaves) return;                 // wave-uniform -> EXEC stays all-1
  const int tm = wave / tilesN;
  const int tn = wave - tm * tilesN;

  const _Float16* Abase = A + (size_t)tm * 16 * K;
  const _Float16* Bbase = Bt + (size_t)tn * 64 * K;

  v8f acc[NT] = {};
#pragma unroll 2
  for (int kk = 0; kk < K; kk += 32) {
    v16h a = load_frag16x32(Abase + kk, K);
#pragma unroll
    for (int c = 0; c < NT; ++c) {
      v16h b = load_frag16x32(Bbase + c * 16 * K + kk, K);
      acc[c] = wmma_f16(a, b, acc[c]);
    }
  }

  const int lane = threadIdx.x & 31;
  const int nl = lane & 15, hi = lane >> 4;   // C/D layout: lane=N, vgpr=M
#pragma unroll
  for (int c = 0; c < NT; ++c) {
    const int n = tn * 64 + c * 16 + nl;
#pragma unroll
    for (int r = 0; r < 8; ++r) {
      const int m = tm * 16 + r + 8 * hi;
      float v = acc[c][r] + bias[n];
      if (addsrc)  v += addsrc[(size_t)m * N + n];
      if (do_gelu) v = 0.5f * v * (1.0f + erff(v * 0.70710678118f));
      if (outF32)  outF32[(size_t)m * N + n] = v;
      if (outF16)  outF16[(size_t)m * N + n] = (_Float16)v;
      if (outF16T) outF16T[(size_t)n * M + m] = (_Float16)v;
    }
  }
}

// ---------------------------------------------------------------------------
// Flash-style masked edge attention.  One wave = one (16-query tile, head).
// Per iter: 32 keys -> 4 WMMAs for S=Q*K^T, adjacency mask + online softmax,
// P transposed through per-wave LDS, 4 WMMAs for O += P*V (V pre-transposed).
// ---------------------------------------------------------------------------
__global__ __launch_bounds__(128) void edge_attn_kernel(
    const _Float16* __restrict__ Q, const _Float16* __restrict__ Kf,
    const _Float16* __restrict__ Vt, const int* __restrict__ srcIdx,
    const int* __restrict__ dstIdx, _Float16* __restrict__ O) {
  __shared__ _Float16 pbuf[4][16 * 32];       // 1KB per wave
  const int w    = threadIdx.x >> 5;
  const int wave = blockIdx.x * 4 + w;
  const int h    = wave & (NHEADS - 1);
  const int e0   = (wave >> 2) << 4;
  const int lane = threadIdx.x & 31;
  const int nl = lane & 15, hi = lane >> 4;

  // Q fragments: hd=64 -> two 16x32 K-chunks, kept resident.
  const _Float16* qbase = Q + (size_t)e0 * DEDGE + h * HDIM;
  const v16h qa0 = load_frag16x32(qbase, DEDGE);
  const v16h qa1 = load_frag16x32(qbase + 32, DEDGE);

  // Endpoints of this lane's 8 query rows (rows m = r + 8*hi, per C layout).
  int se[8], de[8];
#pragma unroll
  for (int r = 0; r < 8; ++r) {
    se[r] = srcIdx[e0 + r + 8 * hi];
    de[r] = dstIdx[e0 + r + 8 * hi];
  }

  float rowM[8], rowS[8];
#pragma unroll
  for (int r = 0; r < 8; ++r) { rowM[r] = -1e30f; rowS[r] = 0.f; }
  v8f acc[4] = {};
  _Float16* pb = &pbuf[w][0];
  const float scale = 0.125f;                 // 1/sqrt(64)

  for (int f0 = 0; f0 < E_EDGES; f0 += 32) {
    // S tiles for two 16-key sub-tiles
    const _Float16* kb = Kf + (size_t)f0 * DEDGE + h * HDIM;
    v8f s0 = {}, s1 = {};
    s0 = wmma_f16(qa0, load_frag16x32(kb, DEDGE), s0);
    s0 = wmma_f16(qa1, load_frag16x32(kb + 32, DEDGE), s0);
    s1 = wmma_f16(qa0, load_frag16x32(kb + 16 * DEDGE, DEDGE), s1);
    s1 = wmma_f16(qa1, load_frag16x32(kb + 16 * DEDGE + 32, DEDGE), s1);

    // Key endpoints for this lane's column
    const int fs0 = srcIdx[f0 + nl],      fd0 = dstIdx[f0 + nl];
    const int fs1 = srcIdx[f0 + 16 + nl], fd1 = dstIdx[f0 + 16 + nl];

#pragma unroll
    for (int r = 0; r < 8; ++r) {
      float v0 = s0[r] * scale, v1 = s1[r] * scale;
      const int es = se[r], ed = de[r];
      const bool adj0 = (es == fs0) | (es == fd0) | (ed == fs0) | (ed == fd0);
      const bool adj1 = (es == fs1) | (es == fd1) | (ed == fs1) | (ed == fd1);
      v0 = adj0 ? v0 : -1e9f;
      v1 = adj1 ? v1 : -1e9f;
      // chunk max over the 16 lanes of this half-wave (xor masks stay in-half)
      float mx = fmaxf(v0, v1);
      mx = fmaxf(mx, __shfl_xor(mx, 1));
      mx = fmaxf(mx, __shfl_xor(mx, 2));
      mx = fmaxf(mx, __shfl_xor(mx, 4));
      mx = fmaxf(mx, __shfl_xor(mx, 8));
      const float newM  = fmaxf(rowM[r], mx);
      const float alpha = __expf(rowM[r] - newM);
      const float p0 = __expf(v0 - newM), p1 = __expf(v1 - newM);
      float ls = p0 + p1;
      ls += __shfl_xor(ls, 1);
      ls += __shfl_xor(ls, 2);
      ls += __shfl_xor(ls, 4);
      ls += __shfl_xor(ls, 8);
      rowS[r] = rowS[r] * alpha + ls;
      rowM[r] = newM;
#pragma unroll
      for (int c = 0; c < 4; ++c) acc[c][r] *= alpha;   // rescale O rows
      // transpose P into LDS: row m = r+8*hi, cols nl and 16+nl
      const int m = r + 8 * hi;
      pb[m * 32 + nl]      = (_Float16)p0;
      pb[m * 32 + 16 + nl] = (_Float16)p1;
    }
    asm volatile("s_wait_dscnt 0" ::: "memory");   // P stores visible to frag reload

    const v16h pa = load_frag16x32(pb, 32);        // P as 16x32 A fragment
#pragma unroll
    for (int c = 0; c < 4; ++c) {
      v16h vb = load_frag16x32(Vt + (size_t)(h * HDIM + c * 16) * E_EDGES + f0,
                               E_EDGES);
      acc[c] = wmma_f16(pa, vb, acc[c]);
    }
  }

  // Normalize and store O (row-major [E, 256], head-concatenated)
  float inv[8];
#pragma unroll
  for (int r = 0; r < 8; ++r) inv[r] = __frcp_rn(rowS[r]);
#pragma unroll
  for (int c = 0; c < 4; ++c) {
#pragma unroll
    for (int r = 0; r < 8; ++r) {
      const float v = acc[c][r] * inv[r];
      O[(size_t)(e0 + r + 8 * hi) * DEDGE + h * HDIM + c * 16 + nl] = (_Float16)v;
    }
  }
}

// ---------------------------------------------------------------------------
// Host orchestration
// ---------------------------------------------------------------------------
extern "C" void kernel_launch(void* const* d_in, const int* in_sizes, int n_in,
                              void* d_out, int out_size, void* d_ws, size_t ws_size,
                              hipStream_t stream) {
  const float* nf = (const float*)d_in[0];
  const float* ef = (const float*)d_in[1];
  const int* ei   = (const int*)d_in[2];
  const int* srcIdx = ei;
  const int* dstIdx = ei + E_EDGES;

  // Workspace layout (bytes)
  char* ws = (char*)d_ws;
  const size_t WSLOT = 256 * 256 * sizeof(_Float16);           // 128KB per weight
  auto Wt = [&](int slot) { return (_Float16*)(ws + slot * WSLOT); };
  _Float16* xcat = (_Float16*)(ws + (size_t)2  * 1048576);     // [4096,256] f16
  float*    xF32 = (float*)   (ws + (size_t)4  * 1048576);     // [4096,256] f32
  _Float16* xF16 = (_Float16*)(ws + (size_t)8  * 1048576);     // [4096,256] f16
  _Float16* Qb   = (_Float16*)(ws + (size_t)10 * 1048576);
  _Float16* Kb   = (_Float16*)(ws + (size_t)12 * 1048576);
  _Float16* VTb  = (_Float16*)(ws + (size_t)14 * 1048576);     // [256,4096] f16
  _Float16* Ob   = (_Float16*)(ws + (size_t)16 * 1048576);
  float*    yF32 = (float*)   (ws + (size_t)18 * 1048576);     // [4096,256] f32
  _Float16* hF16 = (_Float16*)(ws + (size_t)22 * 1048576);     // [4096,256] f16

  // f16 transposed weights: slots 0-4 layer1 (Wn,Wq,Wk,Wv,Wo), 5-9 layer2,
  // 10 cls_W1, 11 cls_W2.
  for (int l = 0; l < 2; ++l) {
    const int base = 3 + l * 10;
    for (int j = 0; j < 5; ++j) {
      const float* W = (const float*)d_in[base + 2 * j];       // Wn,Wq,Wk,Wv,Wo
      transpose_w_kernel<<<(256 * 256 + 255) / 256, 256, 0, stream>>>(
          W, Wt(l * 5 + j), 256, 256);
    }
  }
  transpose_w_kernel<<<(256 * 256 + 255) / 256, 256, 0, stream>>>(
      (const float*)d_in[23], Wt(10), 256, 256);
  transpose_w_kernel<<<(256 * NCLS + 255) / 256, 256, 0, stream>>>(
      (const float*)d_in[25], Wt(11), 256, NCLS);

  // Gather + concat endpoint node features (shared by both layers)
  gather_concat_kernel<<<(E_EDGES * 256) / 256, 256, 0, stream>>>(
      nf, srcIdx, dstIdx, xcat);

  // Full-width GEMM: N=256, K=256, 4 subtiles/wave -> 1024 waves, 256 blocks
  auto gemm256 = [&](const _Float16* A, const _Float16* Bt, const float* bias,
                     const float* add, float* o32, _Float16* o16,
                     _Float16* o16t, int gelu) {
    gemm_wmma_kernel<256, 256, 4><<<(E_EDGES / 16), 128, 0, stream>>>(
        A, Bt, bias, add, o32, o16, o16t, E_EDGES, gelu);
  };

  for (int l = 0; l < 2; ++l) {
    const int base = 3 + l * 10;
    const float* bn = (const float*)d_in[base + 1];
    const float* bq = (const float*)d_in[base + 3];
    const float* bk = (const float*)d_in[base + 5];
    const float* bv = (const float*)d_in[base + 7];
    const float* bo = (const float*)d_in[base + 9];
    const float* efL = (l == 0) ? ef : yF32;

    // x = ef + concat(nf[src],nf[dst]) @ Wn + bn   (f32 master + f16 copy)
    gemm256(xcat, Wt(l * 5 + 0), bn, efL, xF32, xF16, nullptr, 0);
    // Q, K row-major f16 ; V transposed f16
    gemm256(xF16, Wt(l * 5 + 1), bq, nullptr, nullptr, Qb, nullptr, 0);
    gemm256(xF16, Wt(l * 5 + 2), bk, nullptr, nullptr, Kb, nullptr, 0);
    gemm256(xF16, Wt(l * 5 + 3), bv, nullptr, nullptr, nullptr, VTb, 0);
    // masked multi-head attention (flash-style, adjacency computed on the fly)
    edge_attn_kernel<<<(E_EDGES / 16) * NHEADS / 4, 128, 0, stream>>>(
        Qb, Kb, VTb, srcIdx, dstIdx, Ob);
    // y = x + O @ Wo + bo  (f32 for next layer residual, f16 for next GEMM A)
    gemm256(Ob, Wt(l * 5 + 4), bo, xF32, yF32, xF16, nullptr, 0);
  }

  // Classifier: h = gelu(y @ W1 + b1) ; out = h @ W2 + b2
  gemm256(xF16, Wt(10), (const float*)d_in[24], nullptr, nullptr, hF16, nullptr, 1);
  // Final: N=16 -> one subtile per wave, 256 waves, 64 blocks
  gemm_wmma_kernel<NCLS, 256, 1><<<E_EDGES / 16 / 4, 128, 0, stream>>>(
      hF16, Wt(11), (const float*)d_in[26], nullptr, (float*)d_out, nullptr,
      nullptr, E_EDGES, 0);
}
